// ConvIntrinsic_58574763983229
// MI455X (gfx1250) — matrix operand order repaired and checked by reading, not compiled, not run on hardware
//
#include <hip/hip_runtime.h>

typedef __attribute__((ext_vector_type(16))) __bf16 v16bf;
typedef __attribute__((ext_vector_type(8)))  __bf16 v8bf;
typedef __attribute__((ext_vector_type(8)))  float  v8f;

#define NV   50000
#define RR   5
#define AA   8
#define CC   64
#define TT   64
#define OO   8
#define RA   40        // R*A
#define KN   2560      // RA*CC  (neighbor K)
#define KC   2624      // KN + CC (center folded in)
#define KSTEPS 82      // KC / 32
#define COLS 512       // OO*TT
#define CT_STRIDE (KSTEPS * 512)   // bf16 elems between column-tile fragment sets
#define ROWT 16        // vertices per block
#define XSTRIDE 2632   // padded LDS row stride in bf16 elems (5264B, 16B aligned)

static __device__ __forceinline__ __bf16 f2bf(float f) {
  // round-to-nearest-even fp32 -> bf16
  unsigned u = __builtin_bit_cast(unsigned, f);
  unsigned r = u + 0x7FFFu + ((u >> 16) & 1u);
  unsigned short h = (unsigned short)(r >> 16);
  __bf16 b;
  __builtin_memcpy(&b, &h, 2);
  return b;
}

// ---------------------------------------------------------------------------
// Kernel A: fold interp_coeff + rotation roll + center weights into one
// bf16 weight matrix Wc[K=2624][512], stored pre-swizzled as WMMA B fragments:
//   fragment (ct, ks): 32 lanes x 16 bf16; element (k,col):
//     ct = col/16, n = col%16, ks = k/32, kk = k%32
//     lane = (kk>=16)*16 + n, slot = kk%16
// ---------------------------------------------------------------------------
__global__ void pack_weights_kernel(const float* __restrict__ nw,   // (T,R,A,C)
                                    const float* __restrict__ cw,   // (T,1,C)
                                    const float* __restrict__ ic,   // (R,A,RA)
                                    __bf16* __restrict__ wc) {
  const unsigned idx = blockIdx.x * blockDim.x + threadIdx.x;
  if (idx >= KC * COLS) return;
  const unsigned k   = idx >> 9;      // / COLS
  const unsigned col = idx & 511;     // % COLS
  const unsigned o = col >> 6;        // rotation
  const unsigned t = col & 63;        // template
  float v = 0.f;
  if (k < KN) {
    const unsigned m = k >> 6;        // flattened (r,a) source index
    const unsigned c = k & 63;
    #pragma unroll
    for (unsigned r = 0; r < RR; ++r) {
      #pragma unroll
      for (unsigned a = 0; a < AA; ++a) {
        v += ic[(r * AA + a) * RA + m] *
             nw[((t * RR + r) * AA + ((a + o) & 7)) * CC + c];
      }
    }
  } else {
    v = cw[t * CC + (k - KN)];        // center fold, same for every rotation o
  }
  const unsigned ct = col >> 4, n = col & 15;
  const unsigned ks = k >> 5,  kk = k & 31;
  const unsigned lane = ((kk >> 4) << 4) | n;
  const unsigned slot = kk & 15;
  wc[((ct * KSTEPS + ks) * 32 + lane) * 16 + slot] = f2bf(v);
}

// ---------------------------------------------------------------------------
// Kernel B: per block = 16 vertices x 512 output columns.
//   Phase 1: barycentric gather -> bf16 activation tile in LDS (16 x 2624)
//            (lane-group g handles row g; no integer division)
//   Phase 2: 8 waves x 4 column tiles, K-loop of v_wmma_f32_16x16x32_bf16
//   Phase 3: bias + relu + store
// ---------------------------------------------------------------------------
__global__ void __launch_bounds__(256)
conv_wmma_kernel(const float* __restrict__ ms,    // (N,C)
                 const float* __restrict__ bc,    // (N,R,A,3,2)
                 const float* __restrict__ bias,  // (T,)
                 const __bf16* __restrict__ wc,   // swizzled (32,82,32,16)
                 float* __restrict__ out) {       // (N,O,T) = (N,512)
  extern __shared__ __align__(16) __bf16 xs[];    // [ROWT][XSTRIDE]
  const unsigned vt   = blockIdx.x;
  const unsigned tid  = threadIdx.x;
  const unsigned lane = tid & 31;
  const unsigned wave = tid >> 5;

  // ---------------- Phase 1: build activation tile ----------------
  {
    const unsigned g  = tid >> 4;              // 16 groups <-> 16 rows
    const unsigned ln = tid & 15;              // channel lane within group
    const unsigned n  = vt * ROWT + g;         // global vertex for this group
    const float* b6 = bc + (unsigned)(n * (RA * 6));   // (idx,w) x3 per (n,m)
    __bf16* xrow = xs + g * XSTRIDE;
    #pragma unroll 2
    for (unsigned m = 0; m < RA; ++m) {
      const int   i0 = (int)b6[0]; const float w0 = b6[1];
      const int   i1 = (int)b6[2]; const float w1 = b6[3];
      const int   i2 = (int)b6[4]; const float w2 = b6[5];
      b6 += 6;
      const float* r0 = ms + (unsigned)i0 * CC;
      const float* r1 = ms + (unsigned)i1 * CC;
      const float* r2 = ms + (unsigned)i2 * CC;
      #pragma unroll
      for (unsigned cb = 0; cb < 4; ++cb) {
        const unsigned c = cb * 16 + ln;
        const float acc = w0 * r0[c] + w1 * r1[c] + w2 * r2[c];
        xrow[m * CC + c] = f2bf(acc);
      }
    }
    // center channels appended at K = KN..KN+63 (4 channels per thread)
    {
      const unsigned c4 = (tid & 15) * 4;       // 16 threads x 4ch cover a row
      #pragma unroll
      for (unsigned q = 0; q < 4; ++q) {
        const unsigned c = c4 + q;
        xrow[KN + c] = f2bf(ms[n * CC + c]);
      }
    }
  }
  __syncthreads();

  // ---------------- Phase 2: WMMA main loop ----------------
  v8f acc0 = {}, acc1 = {}, acc2 = {}, acc3 = {};
  const unsigned arow  = lane & 15;          // A: M = lane%16
  const unsigned ahalf = (lane >> 4) * 16;   // lanes 16-31 hold K 16-31
  const unsigned ct0   = wave * 4;           // 4 column tiles per wave
  const __bf16* bbase = wc + ct0 * CT_STRIDE + lane * 16;
  const __bf16* arowp = xs + arow * XSTRIDE + ahalf;

  #pragma unroll 2
  for (unsigned ks = 0; ks < KSTEPS; ++ks) {
    // A fragment from LDS (contiguous k = 16h + s, same order as the Wc packing)
    const __bf16* ap = arowp + ks * 32;
    v8bf alo = *(const v8bf*)ap;             // K k0..k0+7
    v8bf ahi = *(const v8bf*)(ap + 8);       // K k0+8..k0+15
    v16bf a = __builtin_shufflevector(alo, ahi,
                0, 1, 2, 3, 4, 5, 6, 7, 8, 9, 10, 11, 12, 13, 14, 15);

    const __bf16* bp = bbase + ks * 512;
    if (ks + 1 < KSTEPS) __builtin_prefetch(bp + 512, 0, 0);
    v16bf b0 = *(const v16bf*)(bp);
    v16bf b1 = *(const v16bf*)(bp + CT_STRIDE);
    v16bf b2 = *(const v16bf*)(bp + 2 * CT_STRIDE);
    v16bf b3 = *(const v16bf*)(bp + 3 * CT_STRIDE);

    acc0 = __builtin_amdgcn_wmma_f32_16x16x32_bf16(false, a, false, b0, (short)0, acc0, false, false);
    acc1 = __builtin_amdgcn_wmma_f32_16x16x32_bf16(false, a, false, b1, (short)0, acc1, false, false);
    acc2 = __builtin_amdgcn_wmma_f32_16x16x32_bf16(false, a, false, b2, (short)0, acc2, false, false);
    acc3 = __builtin_amdgcn_wmma_f32_16x16x32_bf16(false, a, false, b3, (short)0, acc3, false, false);
  }

  // ---------------- Phase 3: bias + relu + store ----------------
  // C/D layout: lanes 0-15 -> N=lane, M=vgpr; lanes 16-31 -> N=lane-16, M=vgpr+8
  const unsigned ncol = lane & 15;
  const unsigned mofs = (lane >> 4) * 8;
  const unsigned rowbase = vt * ROWT + mofs;
  v8f accs[4] = {acc0, acc1, acc2, acc3};
  #pragma unroll
  for (unsigned j = 0; j < 4; ++j) {
    const unsigned col = (ct0 + j) * 16 + ncol;
    const float bb = bias[col & 63];
    float* op = out + rowbase * COLS + col;
    #pragma unroll
    for (unsigned v = 0; v < 8; ++v) {
      float x = accs[j][v] + bb;
      op[v * COLS] = x > 0.f ? x : 0.f;
    }
  }
}

// ---------------------------------------------------------------------------
extern "C" void kernel_launch(void* const* d_in, const int* in_sizes, int n_in,
                              void* d_out, int out_size, void* d_ws, size_t ws_size,
                              hipStream_t stream) {
  const float* ms   = (const float*)d_in[0];   // mesh_signal      (N,C)
  const float* bc   = (const float*)d_in[1];   // bary_coordinates (N,R,A,3,2)
  const float* nw   = (const float*)d_in[2];   // neighbor_weights (T,R,A,C)
  const float* cw   = (const float*)d_in[3];   // center_weights   (T,1,C)
  const float* bias = (const float*)d_in[4];   // bias             (T,)
  const float* ic   = (const float*)d_in[5];   // interp_coeff     (R,A,RA)
  __bf16* wcfrag = (__bf16*)d_ws;              // 2624*512*2 = 2.68 MB

  {
    const int total = KC * COLS;               // 1,343,488
    pack_weights_kernel<<<(total + 255) / 256, 256, 0, stream>>>(nw, cw, ic, wcfrag);
  }

  const size_t smem = (size_t)ROWT * XSTRIDE * sizeof(__bf16);  // 84,224 B
  hipFuncSetAttribute(reinterpret_cast<const void*>(conv_wmma_kernel),
                      hipFuncAttributeMaxDynamicSharedMemorySize, (int)smem);
  conv_wmma_kernel<<<NV / ROWT, 256, smem, stream>>>(ms, bc, bias, wcfrag,
                                                     (float*)d_out);
}
